// MemoryEfficientAttention_62234076119055
// MI455X (gfx1250) — matrix operand order repaired and checked by reading, not compiled, not run on hardware
//
#include <hip/hip_runtime.h>
#include <hip/hip_bf16.h>

typedef __bf16 bf16;
typedef __attribute__((ext_vector_type(16))) __bf16 v16bf;
typedef __attribute__((ext_vector_type(8)))  __bf16 v8bf;
typedef __attribute__((ext_vector_type(8)))  float  v8f;

#define B_   2
#define N_   2048
#define C_   1024
#define H_   16
#define DH_  64
#define SCALE_ 0.125f                 // 64^-0.5
#define QS_  (B_*H_*N_*DH_)           // elements per q/k/v tensor = 4194304
#define MROWS_ (B_*N_)                // 4096

static __device__ __forceinline__ bf16 f2bf(float f) {
  unsigned u = __builtin_bit_cast(unsigned, f);
  unsigned r = u + 0x7FFFu + ((u >> 16) & 1u);   // round-to-nearest-even
  unsigned short h = (unsigned short)(r >> 16);
  return __builtin_bit_cast(bf16, h);
}

template <typename P>
static __device__ __forceinline__ v16bf frag16(P lo, P hi) {
  v8bf a = *(const v8bf*)lo;
  v8bf b = *(const v8bf*)hi;
  return __builtin_shufflevector(a, b, 0,1,2,3,4,5,6,7,8,9,10,11,12,13,14,15);
}

static __device__ __forceinline__ v8f wmma_bf16(v16bf a, v16bf b, v8f c) {
  return __builtin_amdgcn_wmma_f32_16x16x32_bf16(false, a, false, b, (short)0, c,
                                                 false, false);
}

// ---------------- fp32 -> bf16 conversion ----------------
__global__ void cvt_f32_bf16(const float* __restrict__ in, bf16* __restrict__ out, int n) {
  int i = blockIdx.x * blockDim.x + threadIdx.x;
  int stride = gridDim.x * blockDim.x;
  for (; i < n; i += stride) out[i] = f2bf(in[i]);
}

// ---------------- QKV GEMM: [4096,1024] @ W[3072,1024]^T -> q/k/v [B,H,N,64] bf16 ----
// Wave tile: 32 rows x 64 cols (two 16-row A slabs share every B fragment).
__global__ __launch_bounds__(256) void qkv_gemm(const bf16* __restrict__ xb,
                                                const bf16* __restrict__ wb,
                                                bf16* __restrict__ qkv) {
  const int tid  = threadIdx.x;
  const int wave = tid >> 5;
  const int lane = tid & 31;
  const int l16  = lane & 15;
  const int half = lane >> 4;
  const int colbase = blockIdx.x * 64;                 // 64 output features / wave
  const int rowbase = blockIdx.y * 256 + wave * 32;    // 32 rows / wave

  const bf16* arow0 = xb + (size_t)(rowbase + l16) * C_;
  const bf16* arow1 = arow0 + (size_t)16 * C_;
  const bf16* brow[4];
#pragma unroll
  for (int t = 0; t < 4; ++t)
    brow[t] = wb + (size_t)(colbase + t * 16 + l16) * C_;

  v8f acc[2][4] = {};
  for (int kk = 0; kk < C_; kk += 32) {
    // A 16x32 bf16 layout: lane half holds contiguous 16 K values of its row
    v16bf a0 = frag16(arow0 + kk + half * 16, arow0 + kk + half * 16 + 8);
    v16bf a1 = frag16(arow1 + kk + half * 16, arow1 + kk + half * 16 + 8);
#pragma unroll
    for (int t = 0; t < 4; ++t) {
      // B 32x16 layout: lane half holds contiguous 16 K values of its column
      v16bf b = frag16(brow[t] + kk + half * 16, brow[t] + kk + half * 16 + 8);
      acc[0][t] = wmma_bf16(a0, b, acc[0][t]);
      acc[1][t] = wmma_bf16(a1, b, acc[1][t]);
    }
  }

  const int trip = colbase >> 10;          // 0=q 1=k 2=v
  const int h    = (colbase >> 6) & (H_ - 1);
  bf16* dst = qkv + (size_t)trip * QS_;
#pragma unroll
  for (int s = 0; s < 2; ++s) {
#pragma unroll
    for (int t = 0; t < 4; ++t) {
      const int hd = t * 16 + l16;
#pragma unroll
      for (int i = 0; i < 8; ++i) {
        const int rowg = rowbase + s * 16 + i + 8 * half;  // C-layout: M = i + 8*half
        const int bb = rowg >> 11;
        const int nn = rowg & (N_ - 1);
        dst[(((size_t)bb * H_ + h) * N_ + nn) * DH_ + hd] = f2bf(acc[s][t][i]);
      }
    }
  }
}

// ---------------- Flash attention per (b,h), 64 q-rows per block, 4 waves ----------
__global__ __launch_bounds__(128) void attn_fwd(const bf16* __restrict__ q,
                                                const bf16* __restrict__ k,
                                                const bf16* __restrict__ v,
                                                bf16* __restrict__ ctx) {
  __shared__ bf16 Kt[32][64];        // [key][d] row-major  (filled by async DMA)
  __shared__ bf16 Vt[64][32];        // [d][key] transposed
  __shared__ bf16 Pst[4][16][32];    // per-wave P restage [row][key]

  const int tid  = threadIdx.x;
  const int wave = tid >> 5;
  const int lane = tid & 31;
  const int l16  = lane & 15;
  const int half = lane >> 4;
  const int bh     = blockIdx.x;
  const int rowblk = blockIdx.y * 64;

  const bf16* qb = q + (size_t)bh * N_ * DH_;
  const bf16* kb = k + (size_t)bh * N_ * DH_;
  const bf16* vb = v + (size_t)bh * N_ * DH_;

  // Q fragments: 16 rows x 64 d, split into two K=32 chunks, resident in VGPRs
  const bf16* qr = qb + (size_t)(rowblk + wave * 16 + l16) * DH_;
  v16bf qa[2];
#pragma unroll
  for (int kc = 0; kc < 2; ++kc)
    qa[kc] = frag16(qr + kc * 32 + half * 16, qr + kc * 32 + half * 16 + 8);

  v8f o[4] = {};
  float mi[8], li[8];
#pragma unroll
  for (int i = 0; i < 8; ++i) { mi[i] = -3.0e38f; li[i] = 0.f; }

  for (int kt = 0; kt < N_ / 32; ++kt) {
    const int kbase = kt * 32;
    // cooperative tile load: 32 keys x 64 d.
    // K tile: CDNA5 async global->LDS DMA (ASYNCcnt), no VGPR round trip.
    // V tile: manual transposed scatter (async path cannot transpose).
#pragma unroll
    for (int it = 0; it < 2; ++it) {
      const int idx = tid + it * 128;        // 256 x 16B chunks
      const int r  = idx >> 3;
      const int c8 = (idx & 7) * 8;
      unsigned ldsoff = (unsigned)(size_t)&Kt[r][c8];
      const bf16* gsrc = kb + (size_t)(kbase + r) * DH_ + c8;
      asm volatile("global_load_async_to_lds_b128 %0, %1, off"
                   :: "v"(ldsoff), "v"(gsrc) : "memory");
      v8bf vv = *(const v8bf*)(vb + (size_t)(kbase + r) * DH_ + c8);
#pragma unroll
      for (int j = 0; j < 8; ++j) Vt[c8 + j][r] = vv[j];
    }
    asm volatile("s_wait_asynccnt 0" ::: "memory");
    __syncthreads();

    // S = Q @ K^T : two 16x16 tiles (keys 0..15 and 16..31), K-dim = 64 in 2 chunks
    v8f s0 = {}, s1 = {};
#pragma unroll
    for (int dk = 0; dk < 2; ++dk) {
      v16bf b0 = frag16(&Kt[l16][dk * 32 + half * 16],
                        &Kt[l16][dk * 32 + half * 16 + 8]);
      v16bf b1 = frag16(&Kt[16 + l16][dk * 32 + half * 16],
                        &Kt[16 + l16][dk * 32 + half * 16 + 8]);
      s0 = wmma_bf16(qa[dk], b0, s0);
      s1 = wmma_bf16(qa[dk], b1, s1);
    }

    // online softmax; C-layout: VGPR i holds row (i + 8*half), col l16
#pragma unroll
    for (int i = 0; i < 8; ++i) {
      float a0 = s0[i] * SCALE_;
      float a1 = s1[i] * SCALE_;
      float rm = fmaxf(a0, a1);
      rm = fmaxf(rm, __shfl_xor(rm, 1));
      rm = fmaxf(rm, __shfl_xor(rm, 2));
      rm = fmaxf(rm, __shfl_xor(rm, 4));
      rm = fmaxf(rm, __shfl_xor(rm, 8));
      float mnew  = fmaxf(mi[i], rm);
      float alpha = __expf(mi[i] - mnew);
      float p0 = __expf(a0 - mnew);
      float p1 = __expf(a1 - mnew);
      float rs = p0 + p1;
      rs += __shfl_xor(rs, 1);
      rs += __shfl_xor(rs, 2);
      rs += __shfl_xor(rs, 4);
      rs += __shfl_xor(rs, 8);
      li[i] = li[i] * alpha + rs;
      mi[i] = mnew;
      const int m = i + 8 * half;
      Pst[wave][m][l16]      = f2bf(p0);
      Pst[wave][m][16 + l16] = f2bf(p1);
#pragma unroll
      for (int t = 0; t < 4; ++t) o[t][i] *= alpha;
    }

    // O += P @ V : A-frag from LDS restage (C->A layout), B-frag from transposed V
    v16bf pa = frag16(&Pst[wave][l16][half * 16], &Pst[wave][l16][half * 16 + 8]);
#pragma unroll
    for (int t = 0; t < 4; ++t) {
      v16bf bv = frag16(&Vt[t * 16 + l16][half * 16],
                        &Vt[t * 16 + l16][half * 16 + 8]);
      o[t] = wmma_bf16(pa, bv, o[t]);
    }
    __syncthreads();
  }

  // epilogue: normalize, write ctx [B, N, H*64] in bf16 for the proj GEMM
  const int b = bh >> 4;
  const int h = bh & (H_ - 1);
  float inv[8];
#pragma unroll
  for (int i = 0; i < 8; ++i) inv[i] = 1.0f / li[i];
#pragma unroll
  for (int t = 0; t < 4; ++t) {
#pragma unroll
    for (int i = 0; i < 8; ++i) {
      const int rowg = rowblk + wave * 16 + i + 8 * half;
      ctx[((size_t)b * N_ + rowg) * C_ + h * DH_ + t * 16 + l16] =
          f2bf(o[t][i] * inv[i]);
    }
  }
}

// ---------------- Proj GEMM: ctx[4096,1024] @ Wp[1024,1024]^T + bias -> fp32 --------
__global__ __launch_bounds__(256) void proj_gemm(const bf16* __restrict__ ctx,
                                                 const bf16* __restrict__ wb,
                                                 const float* __restrict__ bias,
                                                 float* __restrict__ out) {
  const int tid  = threadIdx.x;
  const int wave = tid >> 5;
  const int lane = tid & 31;
  const int l16  = lane & 15;
  const int half = lane >> 4;
  const int colbase = blockIdx.x * 64;
  const int rowbase = blockIdx.y * 256 + wave * 32;

  const bf16* arow0 = ctx + (size_t)(rowbase + l16) * C_;
  const bf16* arow1 = arow0 + (size_t)16 * C_;
  const bf16* brow[4];
#pragma unroll
  for (int t = 0; t < 4; ++t)
    brow[t] = wb + (size_t)(colbase + t * 16 + l16) * C_;

  v8f acc[2][4] = {};
  for (int kk = 0; kk < C_; kk += 32) {
    v16bf a0 = frag16(arow0 + kk + half * 16, arow0 + kk + half * 16 + 8);
    v16bf a1 = frag16(arow1 + kk + half * 16, arow1 + kk + half * 16 + 8);
#pragma unroll
    for (int t = 0; t < 4; ++t) {
      v16bf b = frag16(brow[t] + kk + half * 16, brow[t] + kk + half * 16 + 8);
      acc[0][t] = wmma_bf16(a0, b, acc[0][t]);
      acc[1][t] = wmma_bf16(a1, b, acc[1][t]);
    }
  }

#pragma unroll
  for (int s = 0; s < 2; ++s) {
#pragma unroll
    for (int t = 0; t < 4; ++t) {
      const int col = colbase + t * 16 + l16;
      const float bs = bias[col];
#pragma unroll
      for (int i = 0; i < 8; ++i) {
        const int rowg = rowbase + s * 16 + i + 8 * half;
        out[(size_t)rowg * C_ + col] = acc[s][t][i] + bs;
      }
    }
  }
}

extern "C" void kernel_launch(void* const* d_in, const int* in_sizes, int n_in,
                              void* d_out, int out_size, void* d_ws, size_t ws_size,
                              hipStream_t stream) {
  const float* x      = (const float*)d_in[0];
  const float* qkv_w  = (const float*)d_in[1];
  const float* proj_w = (const float*)d_in[2];
  const float* proj_b = (const float*)d_in[3];
  float* out = (float*)d_out;

  char* ws = (char*)d_ws;
  bf16* xb    = (bf16*)ws; ws += (size_t)MROWS_ * C_ * 2;   // 8 MB
  bf16* wqkv  = (bf16*)ws; ws += (size_t)3 * C_ * C_ * 2;   // 6 MB
  bf16* wproj = (bf16*)ws; ws += (size_t)C_ * C_ * 2;       // 2 MB
  bf16* qkv   = (bf16*)ws; ws += (size_t)3 * QS_ * 2;       // 24 MB
  bf16* ctx   = (bf16*)ws;                                  // 8 MB

  cvt_f32_bf16<<<1024, 256, 0, stream>>>(x,      xb,    MROWS_ * C_);
  cvt_f32_bf16<<<1024, 256, 0, stream>>>(qkv_w,  wqkv,  3 * C_ * C_);
  cvt_f32_bf16<<<1024, 256, 0, stream>>>(proj_w, wproj, C_ * C_);

  qkv_gemm <<<dim3(48, 16), 256, 0, stream>>>(xb, wqkv, qkv);
  attn_fwd <<<dim3(32, 32), 128, 0, stream>>>(qkv, qkv + QS_, qkv + 2 * QS_, ctx);
  proj_gemm<<<dim3(16, 16), 256, 0, stream>>>(ctx, wproj, proj_b, out);
}
